// DynamicKGE_13297218748557
// MI455X (gfx1250) — matrix operand order repaired and checked by reading, not compiled, not run on hardware
//
#include <hip/hip_runtime.h>
#include <math.h>

typedef __attribute__((ext_vector_type(16))) _Float16 v16h;
typedef __attribute__((ext_vector_type(8)))  float    v8f;

#define AS1 __attribute__((address_space(1)))
#define AS3 __attribute__((address_space(3)))

// Pointee types matching the builtin signatures exactly (GCC vector_size style).
typedef int async_v2 __attribute__((vector_size(8)));   // b64 chunk
typedef int async_v4 __attribute__((vector_size(16)));  // b128 chunk

#if defined(__has_builtin)
#if __has_builtin(__builtin_amdgcn_global_load_async_to_lds_b64) && \
    __has_builtin(__builtin_amdgcn_global_load_async_to_lds_b128) && \
    __has_builtin(__builtin_amdgcn_s_wait_asynccnt)
#define USE_ASYNC_LDS 1
#endif
#endif

constexpr int DIM  = 128;
constexpr int MCTX = 10;
constexpr int M1   = 11;
constexpr int TREL = 202;                 // 2R+1 rows + zero row
constexpr int B    = 1024;
constexpr int ENT_ROWS = 4 * B * M1;      // 45056 rows of H (f16)
constexpr int REL_ROWS = 2 * B * M1;      // 22528 rows of H_rel (f32)
constexpr int MAXE     = 4 * B * M1 * M1; // 495616 potential edges
constexpr int TILES_Y  = 256;             // 16-row tiles per type (cap 4096 edges/type)

struct EntPtrs {
  const int*   center[4];
  const int*   adj[4];
  const int*   Rm[4];
  const int*   nn[4];
  const float* Dm[4];
};

// ---------------- utility kernels ----------------

__global__ void zero_f32(float* p, int n) {
  int i = blockIdx.x * blockDim.x + threadIdx.x;
  int stride = gridDim.x * blockDim.x;
  for (; i < n; i += stride) p[i] = 0.f;
}

__global__ void convert_weights(const float* __restrict__ egw, const float* __restrict__ rgw,
                                _Float16* __restrict__ Wh, _Float16* __restrict__ Wrelh) {
  int i = blockIdx.x * blockDim.x + threadIdx.x;
  int stride = gridDim.x * blockDim.x;
  const int NW = TREL * DIM * DIM;        // incl. zero row
  const int NE = (TREL - 1) * DIM * DIM;  // real weights
  for (int k = i; k < NW; k += stride)
    Wh[k] = (k < NE) ? (_Float16)egw[k] : (_Float16)0.f;
  for (int k = i; k < DIM * DIM; k += stride)
    Wrelh[k] = (_Float16)rgw[k];
}

// H_ent[g,b,0] = entity_emb[center]; H_ent[g,b,1+m] = entity_context[adj[b][m]]  (f16)
__global__ void build_hent(EntPtrs p, const float* __restrict__ entity_emb,
                           const float* __restrict__ entity_ctx, _Float16* __restrict__ Hent) {
  int row = blockIdx.x;
  int d = threadIdx.x;
  int g = row / (B * M1);
  int rem = row % (B * M1);
  int b = rem / M1, j = rem % M1;
  float v;
  if (j == 0) {
    v = entity_emb[(size_t)p.center[g][b] * DIM + d];
  } else {
    int idx = p.adj[g][b * MCTX + (j - 1)];
    v = entity_ctx[(size_t)idx * DIM + d];
  }
  Hent[(size_t)row * DIM + d] = (_Float16)v;
}

// H_rel[g,b,0] = relation_emb[center]; H_rel[g,b,1+m] = rc[adj[b][m][0]] + rc[adj[b][m][1]]  (f32)
__global__ void build_hrel(const int* __restrict__ pos_r, const int* __restrict__ neg_r,
                           const int* __restrict__ pr_adj, const int* __restrict__ nr_adj,
                           const float* __restrict__ relation_emb,
                           const float* __restrict__ relation_ctx, float* __restrict__ Hrel) {
  int row = blockIdx.x;
  int d = threadIdx.x;
  int g = row / (B * M1);
  int rem = row % (B * M1);
  int b = rem / M1, j = rem % M1;
  const int* adj = (g == 0) ? pr_adj : nr_adj;
  const int* cen = (g == 0) ? pos_r : neg_r;
  float v;
  if (j == 0) {
    v = relation_emb[(size_t)cen[b] * DIM + d];
  } else {
    int i0 = adj[(b * MCTX + (j - 1)) * 2 + 0];
    int i1 = adj[(b * MCTX + (j - 1)) * 2 + 1];
    v = relation_ctx[(size_t)i0 * DIM + d] + relation_ctx[(size_t)i1 * DIM + d];
  }
  Hrel[(size_t)row * DIM + d] = v;
}

// ---------------- edge bucketing by relation type ----------------

__global__ void count_edges(EntPtrs p, int* __restrict__ cnt) {
  int idx = blockIdx.x * blockDim.x + threadIdx.x;
  if (idx >= MAXE) return;
  int g = idx / (B * M1 * M1);
  int rem = idx % (B * M1 * M1);
  int b = rem / (M1 * M1);
  int jk = rem % (M1 * M1);
  int j = jk / M1, k = jk % M1;
  int nn = p.nn[g][b];
  if (j > nn || k > nn) return;
  int t = p.Rm[g][(size_t)b * M1 * M1 + jk];
  if (t >= TREL - 1) return;  // zero-weight row -> contributes nothing
  atomicAdd(&cnt[t], 1);
}

__global__ void scan_offsets(const int* __restrict__ cnt, int* __restrict__ offs,
                             int* __restrict__ fill) {
  if (threadIdx.x == 0 && blockIdx.x == 0) {
    int acc = 0;
    for (int t = 0; t < TREL; ++t) { offs[t] = acc; acc += cnt[t]; fill[t] = 0; }
    offs[TREL] = acc;
  }
}

__global__ void scatter_edges(EntPtrs p, const int* __restrict__ offs, int* __restrict__ fill,
                              int* __restrict__ eRow, float* __restrict__ eD) {
  int idx = blockIdx.x * blockDim.x + threadIdx.x;
  if (idx >= MAXE) return;
  int g = idx / (B * M1 * M1);
  int rem = idx % (B * M1 * M1);
  int b = rem / (M1 * M1);
  int jk = rem % (M1 * M1);
  int j = jk / M1, k = jk % M1;
  int nn = p.nn[g][b];
  if (j > nn || k > nn) return;
  int t = p.Rm[g][(size_t)b * M1 * M1 + jk];
  if (t >= TREL - 1) return;
  int pos = offs[t] + atomicAdd(&fill[t], 1);
  eRow[pos] = (g * B + b) * M1 + j;                 // output/H row index
  eD[pos]   = p.Dm[g][(size_t)b * M1 * M1 + jk];    // per-edge scale
}

// ---------------- R-GCN dense GEMM per relation type (WMMA) ----------------
// block = (type t, 16-edge tile); 8 waves each own a 16-column slab of N=128.
__global__ void rgcn_gemm(const _Float16* __restrict__ Wh, const _Float16* __restrict__ Hent,
                          const int* __restrict__ offs, const int* __restrict__ eRow,
                          const float* __restrict__ eD, float* __restrict__ acc) {
  int t = blockIdx.x;
  int start = offs[t], cnt = offs[t + 1] - start;
  int tileBase = blockIdx.y * 16;
  if (tileBase >= cnt) return;

  __shared__ __align__(16) _Float16 Wsh[128][132];  // 66-dword stride: conflict-free B-frag reads
  __shared__ __align__(16) _Float16 Ash[16][136];   // 272B rows: 16B-aligned for async b128
  __shared__ int   s_row[16];
  __shared__ float s_d[16];

  int tid = threadIdx.x;
  const _Float16* Wp = Wh + (size_t)t * DIM * DIM;
  __builtin_prefetch(Wp, 0, 1);  // gfx1250 global_prefetch path

  // ---- stage W[t] (32 KB) into LDS ----
#ifdef USE_ASYNC_LDS
  // async copy: 4096 x b64 chunks, VGPR-free, tracked by ASYNCcnt
  for (int i = tid; i < DIM * DIM / 4; i += 256) {
    int r = i >> 5, c = (i & 31) * 4;
    __builtin_amdgcn_global_load_async_to_lds_b64(
        (AS1 async_v2*)(void*)(Wp + r * DIM + c),
        (AS3 async_v2*)(void*)&Wsh[r][c], 0, 0);
  }
#else
  for (int i = tid; i < DIM * DIM / 8; i += 256) {
    int r = (i * 8) >> 7, c = (i * 8) & 127;
#pragma unroll
    for (int u = 0; u < 8; ++u) Wsh[r][c + u] = Wp[r * DIM + c + u];
  }
#endif
  if (tid < 16) {
    int e = tileBase + tid;
    if (e < cnt) { s_row[tid] = eRow[start + e]; s_d[tid] = eD[start + e]; }
    else         { s_row[tid] = -1;              s_d[tid] = 0.f; }
  }
  __syncthreads();

  // ---- gather 16 A rows (f16 H rows) into LDS ----
  {
    int r = tid >> 4;
    int c0 = (tid & 15) * 8;
    int row = s_row[r];
    // pad rows read from the all-zero weight row -> zeros, branch-free
    const _Float16* src =
        ((row >= 0) ? (Hent + (size_t)row * DIM) : (Wh + (size_t)(TREL - 1) * DIM * DIM)) + c0;
#ifdef USE_ASYNC_LDS
    __builtin_amdgcn_global_load_async_to_lds_b128(
        (AS1 async_v4*)(void*)src, (AS3 async_v4*)(void*)&Ash[r][c0], 0, 0);
#else
#pragma unroll
    for (int u = 0; u < 8; ++u) Ash[r][c0 + u] = src[u];
#endif
  }
#ifdef USE_ASYNC_LDS
  __builtin_amdgcn_s_wait_asynccnt(0);
#endif
  __syncthreads();

  int wave = tid >> 5, lane = tid & 31;
  int colBase = wave * 16;
  int mrow = lane & 15;
  int kb = (lane >> 4) * 8;   // A frag: lanes 0-15 -> K 0..7/16..23, lanes 16-31 -> K 8..15/24..31
  v8f c = {};
#pragma unroll
  for (int kk = 0; kk < DIM; kk += 32) {
    v16h a, bf;
#pragma unroll
    for (int u = 0; u < 8; ++u) {
      a[u]     = Ash[mrow][kk + kb + u];
      a[u + 8] = Ash[mrow][kk + kb + 16 + u];
    }
    int krow = kk + kb;       // B frag: lane = N, K in same order as A frag
#pragma unroll
    for (int u = 0; u < 16; ++u) bf[u] = Wsh[krow + u + (u & 8)][colBase + mrow];
    c = __builtin_amdgcn_wmma_f32_16x16x32_f16(false, a, false, bf, (short)0, c, false, false);
  }

  // ---- D-scaled scatter-accumulate, branch-free (pads hit dummy row with scale 0) ----
  int n = colBase + (lane & 15);
  int mb = (lane >> 4) * 8;
#pragma unroll
  for (int r8 = 0; r8 < 8; ++r8) {
    int m = mb + r8;
    int row = s_row[m];
    int dstRow = (row >= 0) ? row : ENT_ROWS;  // dummy row, never read
    atomicAdd(&acc[(size_t)dstRow * DIM + n], s_d[m] * c[r8]);
  }
}

// ---------------- relation GCN: relu( (A @ H) @ Wrel ) per (g,b) ----------------
__global__ void relgcn(const float* __restrict__ Hrel, const float* __restrict__ prA,
                       const float* __restrict__ nrA, const _Float16* __restrict__ Wrelh,
                       float* __restrict__ vrel) {
  int gb = blockIdx.x;
  int g = gb >> 10, b = gb & 1023;
  const float* A = ((g == 0) ? prA : nrA) + (size_t)b * M1 * M1;
  const float* H = Hrel + (size_t)(g * B + b) * M1 * DIM;

  __shared__ float    s_A[M1 * M1];
  __shared__ float    Hsh[M1][DIM];
  __shared__ __align__(16) _Float16 Ash[16][136];
  __shared__ __align__(16) _Float16 Wsh[128][132];

  int tid = threadIdx.x;
  // ---- stage Wrel (32 KB) into LDS ----
#ifdef USE_ASYNC_LDS
  for (int i = tid; i < DIM * DIM / 4; i += 256) {
    int r = i >> 5, c = (i & 31) * 4;
    __builtin_amdgcn_global_load_async_to_lds_b64(
        (AS1 async_v2*)(void*)(Wrelh + r * DIM + c),
        (AS3 async_v2*)(void*)&Wsh[r][c], 0, 0);
  }
#else
  for (int i = tid; i < DIM * DIM; i += 256) Wsh[i >> 7][i & 127] = Wrelh[i];
#endif
  if (tid < M1 * M1) s_A[tid] = A[tid];
  for (int i = tid; i < M1 * DIM; i += 256) Hsh[i >> 7][i & 127] = H[i];
  __syncthreads();

  // T1 = A @ H  (11x128, padded to 16 rows), stored f16
  for (int i = tid; i < 16 * DIM; i += 256) {
    int j = i >> 7, d = i & 127;
    float s = 0.f;
    if (j < M1) {
#pragma unroll
      for (int k = 0; k < M1; ++k) s += s_A[j * M1 + k] * Hsh[k][d];
    }
    Ash[j][d] = (_Float16)s;
  }
#ifdef USE_ASYNC_LDS
  __builtin_amdgcn_s_wait_asynccnt(0);
#endif
  __syncthreads();

  int wave = tid >> 5, lane = tid & 31;
  int colBase = wave * 16;
  int mrow = lane & 15;
  int kb = (lane >> 4) * 8;
  v8f c = {};
#pragma unroll
  for (int kk = 0; kk < DIM; kk += 32) {
    v16h a, bf;
#pragma unroll
    for (int u = 0; u < 8; ++u) {
      a[u]     = Ash[mrow][kk + kb + u];
      a[u + 8] = Ash[mrow][kk + kb + 16 + u];
    }
    int krow = kk + kb;
#pragma unroll
    for (int u = 0; u < 16; ++u) bf[u] = Wsh[krow + u + (u & 8)][colBase + mrow];
    c = __builtin_amdgcn_wmma_f32_16x16x32_f16(false, a, false, bf, (short)0, c, false, false);
  }

  int n = colBase + (lane & 15);
  int mb = (lane >> 4) * 8;
#pragma unroll
  for (int r8 = 0; r8 < 8; ++r8) {
    int m = mb + r8;
    if (m < M1)
      vrel[((size_t)(g * B + b) * M1 + m) * DIM + n] = fmaxf(c[r8], 0.f);
  }
}

// ---------------- finalize: attention subgraph + gate + score ----------------

__device__ __forceinline__ float block_sum(float v, float* red) {
  int d = threadIdx.x;
  red[d] = v;
  __syncthreads();
  for (int s = 64; s > 0; s >>= 1) {
    if (d < s) red[d] += red[d + s];
    __syncthreads();
  }
  float r = red[0];
  __syncthreads();
  return r;
}

__device__ float gated_subgraph(const float* __restrict__ V, float centerD, float vvecD,
                                float gateD, float* red, float* s_t) {
  int d = threadIdx.x;
  for (int j = 0; j < M1; ++j) {
    float val = fmaxf(V[j * DIM + d], 0.f);      // ReLU of accumulated GCN output
    float p = fmaxf(val * centerD, 0.f) * vvecD; // tmp = relu(v * o) . vvec
    float s = block_sum(p, red);
    if (d == 0) s_t[j] = s;
  }
  __syncthreads();
  float mx = s_t[0];
  for (int j = 1; j < M1; ++j) mx = fmaxf(mx, s_t[j]);
  float den = 0.f;
  for (int j = 0; j < M1; ++j) den += __expf(s_t[j] - mx);
  float sg = 0.f;
  for (int j = 0; j < M1; ++j) {
    float a = __expf(s_t[j] - mx) / den;
    sg += a * fmaxf(V[j * DIM + d], 0.f);
  }
  __syncthreads();
  return gateD * centerD + (1.f - gateD) * sg;
}

__global__ void finalize_kernel(const float* __restrict__ acc, const float* __restrict__ vrel,
                                const float* __restrict__ entity_emb,
                                const float* __restrict__ relation_emb,
                                const int* pos_h, const int* pos_r, const int* pos_t,
                                const int* neg_h, const int* neg_r, const int* neg_t,
                                const float* gate_e, const float* gate_r,
                                const float* v_ent, const float* v_rel, float* __restrict__ out) {
  __shared__ float red[DIM];
  __shared__ float s_t[M1];
  int b = blockIdx.x, d = threadIdx.x;
  float ge = 1.f / (1.f + __expf(-gate_e[d]));
  float gr = 1.f / (1.f + __expf(-gate_r[d]));
  float ve = v_ent[d], vr = v_rel[d];

  const float* Vph = acc + (size_t)(0 * B + b) * M1 * DIM;
  const float* Vpt = acc + (size_t)(1 * B + b) * M1 * DIM;
  const float* Vnh = acc + (size_t)(2 * B + b) * M1 * DIM;
  const float* Vnt = acc + (size_t)(3 * B + b) * M1 * DIM;
  const float* Vpr = vrel + (size_t)(0 * B + b) * M1 * DIM;
  const float* Vnr = vrel + (size_t)(1 * B + b) * M1 * DIM;

  float c_ph = entity_emb[(size_t)pos_h[b] * DIM + d];
  float c_pt = entity_emb[(size_t)pos_t[b] * DIM + d];
  float c_nh = entity_emb[(size_t)neg_h[b] * DIM + d];
  float c_nt = entity_emb[(size_t)neg_t[b] * DIM + d];
  float c_pr = relation_emb[(size_t)pos_r[b] * DIM + d];
  float c_nr = relation_emb[(size_t)neg_r[b] * DIM + d];

  float ph_o = gated_subgraph(Vph, c_ph, ve, ge, red, s_t);
  float pt_o = gated_subgraph(Vpt, c_pt, ve, ge, red, s_t);
  float nh_o = gated_subgraph(Vnh, c_nh, ve, ge, red, s_t);
  float nt_o = gated_subgraph(Vnt, c_nt, ve, ge, red, s_t);
  float pr_o = gated_subgraph(Vpr, c_pr, vr, gr, red, s_t);
  float nr_o = gated_subgraph(Vnr, c_nr, vr, gr, red, s_t);

  float pd = ph_o + pr_o - pt_o;
  float ps = block_sum(pd * pd, red);
  float nd = nh_o + nr_o - nt_o;
  float ns = block_sum(nd * nd, red);
  if (d == 0) {
    out[b]     = sqrtf(ps);
    out[B + b] = sqrtf(ns);
  }
}

// ---------------- host launch ----------------

extern "C" void kernel_launch(void* const* d_in, const int* in_sizes, int n_in,
                              void* d_out, int out_size, void* d_ws, size_t ws_size,
                              hipStream_t stream) {
  const float* entity_emb   = (const float*)d_in[0];
  const float* relation_emb = (const float*)d_in[1];
  const float* entity_ctx   = (const float*)d_in[2];
  const float* relation_ctx = (const float*)d_in[3];
  const float* egw          = (const float*)d_in[4];
  const float* rgw          = (const float*)d_in[5];
  const float* gate_e       = (const float*)d_in[6];
  const float* gate_r       = (const float*)d_in[7];
  const float* v_ent        = (const float*)d_in[8];
  const float* v_rel        = (const float*)d_in[9];
  const int* pos_h = (const int*)d_in[10];
  const int* pos_r = (const int*)d_in[11];
  const int* pos_t = (const int*)d_in[12];
  const int* neg_h = (const int*)d_in[13];
  const int* neg_r = (const int*)d_in[14];
  const int* neg_t = (const int*)d_in[15];
  const int* ph_adj = (const int*)d_in[16];
  const int* pt_adj = (const int*)d_in[17];
  const int* nh_adj = (const int*)d_in[18];
  const int* nt_adj = (const int*)d_in[19];
  const int* pr_adj = (const int*)d_in[20];
  const int* nr_adj = (const int*)d_in[21];
  const int* ph_R = (const int*)d_in[22];
  const int* pt_R = (const int*)d_in[23];
  const int* nh_R = (const int*)d_in[24];
  const int* nt_R = (const int*)d_in[25];
  const int* ph_nn = (const int*)d_in[26];
  const int* pt_nn = (const int*)d_in[27];
  const int* nh_nn = (const int*)d_in[28];
  const int* nt_nn = (const int*)d_in[29];
  const float* ph_D = (const float*)d_in[30];
  const float* pt_D = (const float*)d_in[31];
  const float* nh_D = (const float*)d_in[32];
  const float* nt_D = (const float*)d_in[33];
  const float* pr_A = (const float*)d_in[34];
  const float* nr_A = (const float*)d_in[35];
  (void)in_sizes; (void)n_in; (void)out_size; (void)ws_size;

  char* ws = (char*)d_ws;
  size_t off = 0;
  auto carve = [&](size_t bytes) -> char* {
    off = (off + 255) & ~(size_t)255;
    char* p = ws + off;
    off += bytes;
    return p;
  };
  _Float16* Wh    = (_Float16*)carve((size_t)TREL * DIM * DIM * 2);       // ~6.6 MB
  _Float16* Wrelh = (_Float16*)carve((size_t)DIM * DIM * 2);
  _Float16* Hent  = (_Float16*)carve((size_t)ENT_ROWS * DIM * 2);         // ~11.5 MB
  float*    Hrel  = (float*)carve((size_t)REL_ROWS * DIM * 4);            // ~11.5 MB
  float*    acc   = (float*)carve((size_t)(ENT_ROWS + 1) * DIM * 4);      // ~23 MB (+dummy row)
  float*    vrel  = (float*)carve((size_t)REL_ROWS * DIM * 4);            // ~11.5 MB
  int*      cnt   = (int*)carve((TREL + 2) * 4);
  int*      offs  = (int*)carve((TREL + 2) * 4);
  int*      fill  = (int*)carve((TREL + 2) * 4);
  int*      eRow  = (int*)carve((size_t)MAXE * 4);
  float*    eD    = (float*)carve((size_t)MAXE * 4);

  EntPtrs ep;
  ep.center[0] = pos_h;  ep.center[1] = pos_t;  ep.center[2] = neg_h;  ep.center[3] = neg_t;
  ep.adj[0] = ph_adj;    ep.adj[1] = pt_adj;    ep.adj[2] = nh_adj;    ep.adj[3] = nt_adj;
  ep.Rm[0]  = ph_R;      ep.Rm[1]  = pt_R;      ep.Rm[2]  = nh_R;      ep.Rm[3]  = nt_R;
  ep.nn[0]  = ph_nn;     ep.nn[1]  = pt_nn;     ep.nn[2]  = nh_nn;     ep.nn[3]  = nt_nn;
  ep.Dm[0]  = ph_D;      ep.Dm[1]  = pt_D;      ep.Dm[2]  = nh_D;      ep.Dm[3]  = nt_D;

  zero_f32<<<2048, 256, 0, stream>>>(acc, (ENT_ROWS + 1) * DIM);
  zero_f32<<<1, 256, 0, stream>>>((float*)cnt, TREL + 2);
  convert_weights<<<2048, 256, 0, stream>>>(egw, rgw, Wh, Wrelh);
  build_hent<<<ENT_ROWS, 128, 0, stream>>>(ep, entity_emb, entity_ctx, Hent);
  build_hrel<<<REL_ROWS, 128, 0, stream>>>(pos_r, neg_r, pr_adj, nr_adj,
                                           relation_emb, relation_ctx, Hrel);
  count_edges<<<(MAXE + 255) / 256, 256, 0, stream>>>(ep, cnt);
  scan_offsets<<<1, 1, 0, stream>>>(cnt, offs, fill);
  scatter_edges<<<(MAXE + 255) / 256, 256, 0, stream>>>(ep, offs, fill, eRow, eD);
  rgcn_gemm<<<dim3(TREL, TILES_Y), 256, 0, stream>>>(Wh, Hent, offs, eRow, eD, acc);
  relgcn<<<2 * B, 256, 0, stream>>>(Hrel, pr_A, nr_A, Wrelh, vrel);
  finalize_kernel<<<B, 128, 0, stream>>>(acc, vrel, entity_emb, relation_emb,
                                         pos_h, pos_r, pos_t, neg_h, neg_r, neg_t,
                                         gate_e, gate_r, v_ent, v_rel, (float*)d_out);
}
